// M3GNet_45887430590899
// MI455X (gfx1250) — matrix operand order repaired and checked by reading, not compiled, not run on hardware
//
#include <hip/hip_runtime.h>
#include <hip/hip_bf16.h>
#include <math.h>

#define N_ATOMS   50000
#define N_BONDS   1000000
#define N_TRIPLES 1500000
#define N_GRAPHS  64
#define UNITS     64
#define RBF       48
#define SHF       16
#define CUTOFF_R  5.0f

typedef __attribute__((ext_vector_type(16))) __bf16 v16bf;
typedef __attribute__((ext_vector_type(8)))  float  v8f;

union FragU { uint4 q[2]; v16bf v; };

__device__ __forceinline__ unsigned short f2bf_bits(float x) {
  unsigned u = __float_as_uint(x);
  unsigned r = u + 0x7FFFu + ((u >> 16) & 1u);   // round-to-nearest-even
  return (unsigned short)(r >> 16);
}
__device__ __forceinline__ float bf2f(unsigned short b) {
  return __uint_as_float(((unsigned)b) << 16);
}
// native v_exp_f32 / v_rcp_f32 based activations
__device__ __forceinline__ float fast_sigmoid(float x) {
  return __builtin_amdgcn_rcpf(1.0f + __expf(-x));
}
// silu(x1) * sigmoid(x2) = x1 / ((1+e^-x1)(1+e^-x2)) : 2 exp + 1 rcp
__device__ __forceinline__ float fused_gate(float x1, float x2) {
  float d = (1.0f + __expf(-x1)) * (1.0f + __expf(-x2));
  return x1 * __builtin_amdgcn_rcpf(d);
}
__device__ __forceinline__ float fast_silu(float x) { return x * fast_sigmoid(x); }

// ---------------- geometry: bond vectors + lengths ----------------
__global__ void geom_kernel(const float* __restrict__ pos,
                            const float* __restrict__ pbc,
                            const float* __restrict__ lat,
                            const int*   __restrict__ bidx,
                            const int*   __restrict__ batch_bond,
                            float* __restrict__ vec,
                            float* __restrict__ rlen) {
  int b = blockIdx.x * blockDim.x + threadIdx.x;
  if (b >= N_BONDS) return;
  int s = bidx[2*b], r = bidx[2*b+1];
  const float* L = lat + (size_t)batch_bond[b] * 9;
  float o0 = pbc[3*b], o1 = pbc[3*b+1], o2 = pbc[3*b+2];
  float acc = 0.f;
  #pragma unroll
  for (int j = 0; j < 3; ++j) {
    float x = pos[r*3+j] - pos[s*3+j] + o0*L[j] + o1*L[3+j] + o2*L[6+j];
    vec[3*b+j] = x;
    acc += x * x;
  }
  rlen[b] = sqrtf(acc);
}

// ---------------- initial bond / atom features ----------------
__global__ void bond_init_kernel(const float* __restrict__ rlen,
                                 const float* __restrict__ Wb,
                                 const float* __restrict__ bb,
                                 float* __restrict__ bond_feat) {
  int idx = blockIdx.x * blockDim.x + threadIdx.x;
  if (idx >= N_BONDS * UNITS) return;
  int b = idx >> 6, u = idx & 63;
  bond_feat[idx] = fast_silu(rlen[b] * Wb[u] + bb[u]);
}

__global__ void atom_init_kernel(const float* __restrict__ emb,
                                 const int*   __restrict__ types,
                                 float* __restrict__ atom_feat) {
  int idx = blockIdx.x * blockDim.x + threadIdx.x;
  if (idx >= N_ATOMS * UNITS) return;
  int a = idx >> 6, u = idx & 63;
  atom_feat[idx] = emb[(size_t)types[a] * UNITS + u];
}

// ---------------- three-body basis (radial x Chebyshev), stored bf16 ----------------
__global__ void basis_kernel(const float* __restrict__ vec,
                             const int*   __restrict__ trip,
                             const int*   __restrict__ bidx,
                             unsigned short* __restrict__ basis,
                             int* __restrict__ end_atom) {
  int t = blockIdx.x * blockDim.x + threadIdx.x;
  if (t >= N_TRIPLES) return;
  int b1 = trip[2*t], b2 = trip[2*t+1];
  float v1x = vec[3*b1], v1y = vec[3*b1+1], v1z = vec[3*b1+2];
  float v2x = vec[3*b2], v2y = vec[3*b2+1], v2z = vec[3*b2+2];
  float n1 = sqrtf(v1x*v1x + v1y*v1y + v1z*v1z);
  float n2 = sqrtf(v2x*v2x + v2y*v2y + v2z*v2z);
  float inv1 = __builtin_amdgcn_rcpf(n1 + 1e-8f);
  float inv2 = __builtin_amdgcn_rcpf(n2 + 1e-8f);
  float cost = (v1x*v2x + v1y*v2y + v1z*v2z) * inv1 * inv2;
  cost = fminf(1.f, fmaxf(-1.f, cost));
  float r2  = n2;
  float r2s = r2 + 1e-8f;
  float u  = fminf(1.f, fmaxf(0.f, r2 * (1.0f / CUTOFF_R)));
  float u2 = u*u, u3 = u2*u;
  float fc = 1.f - 6.f*u3*u2 + 15.f*u2*u2 - 10.f*u3;
  float inv_r2s = __builtin_amdgcn_rcpf(r2s);
  float rad[3];
  #pragma unroll
  for (int n = 0; n < 3; ++n)
    rad[n] = __sinf((float)(n+1) * 3.14159265358979323846f * r2s * (1.0f / CUTOFF_R))
             * inv_r2s * fc;
  float cheb[SHF];
  cheb[0] = 1.f; cheb[1] = cost;
  #pragma unroll
  for (int l = 2; l < SHF; ++l) cheb[l] = 2.f * cost * cheb[l-1] - cheb[l-2];
  unsigned short* out = basis + (size_t)t * RBF;
  #pragma unroll
  for (int n = 0; n < 3; ++n)
    #pragma unroll
    for (int l = 0; l < SHF; ++l)
      out[n*SHF + l] = f2bf_bits(rad[n] * cheb[l]);
  end_atom[t] = bidx[2*b2 + 1];
}

// ---------------- gate = sigmoid(atom_feat @ Wup + bup) ----------------
__global__ void gate_kernel(const float* __restrict__ atom_feat,
                            const float* __restrict__ Wup,
                            const float* __restrict__ bup,
                            float* __restrict__ gate) {
  int idx = blockIdx.x * blockDim.x + threadIdx.x;
  if (idx >= N_ATOMS * RBF) return;
  int a = idx / RBF, j = idx % RBF;
  float acc = bup[j];
  #pragma unroll 8
  for (int u = 0; u < UNITS; ++u)
    acc += atom_feat[(size_t)a*UNITS + u] * Wup[u*RBF + j];
  gate[idx] = fast_sigmoid(acc);
}

// ---------------- weighted basis scatter-add into agg ----------------
__global__ void scatter_kernel(const unsigned short* __restrict__ basis,
                               const float* __restrict__ gate,
                               const int*   __restrict__ trip,
                               const int*   __restrict__ end_atom,
                               float* __restrict__ agg) {
  int idx = blockIdx.x * blockDim.x + threadIdx.x;
  if (idx >= N_TRIPLES * RBF) return;
  int t = idx / RBF, j = idx % RBF;
  if (t + 2048 < N_TRIPLES)  // prefetch basis ahead -> global_prefetch_b8
    __builtin_prefetch(basis + (size_t)(t + 2048) * RBF + j, 0, 0);
  float w = bf2f(basis[idx]) * gate[(size_t)end_atom[t] * RBF + j];
  atomicAdd(&agg[(size_t)trip[2*t] * RBF + j], w);
}

// ---------------- fused dual gated GEMM via WMMA bf16 ----------------
// MODE 0: A = agg (M x 48, padded to 64);   bond_feat += silu(.)*sigmoid(.)
// MODE 1: A = [atom[send] | atom[recv] | bond_feat] (M x 192); bond_feat += ...
// MODE 2: A = bond_feat (M x 64); atomicAdd into atom_feat[recv]
template<int MODE, int K, int KPAD>
__global__ __launch_bounds__(256) void dual_gemm_kernel(
    const float* Asrc, const float* atom_ro, const int* __restrict__ bidx,
    const float* __restrict__ W1, const float* __restrict__ b1g,
    const float* __restrict__ W2, const float* __restrict__ b2g,
    float* bond_feat, float* atom_wr) {
  __shared__ unsigned short As [32 * KPAD];
  __shared__ unsigned short Bs1[64 * KPAD];   // stored transposed: [n][k]
  __shared__ unsigned short Bs2[64 * KPAD];
  __shared__ float sb1[64], sb2[64];

  const int tid  = threadIdx.x;
  const int row0 = blockIdx.x * 32;

  // stage weights (transposed, K padded with zeros) as bf16
  for (int i = tid; i < 64 * KPAD; i += 256) {
    int n = i / KPAD, k = i % KPAD;
    float w1 = (k < K) ? W1[k*64 + n] : 0.f;
    float w2 = (k < K) ? W2[k*64 + n] : 0.f;
    Bs1[i] = f2bf_bits(w1);
    Bs2[i] = f2bf_bits(w2);
  }
  if (tid < 64) { sb1[tid] = b1g[tid]; sb2[tid] = b2g[tid]; }

  // stage A tile (32 rows x KPAD) as bf16
  for (int i = tid; i < 32 * KPAD; i += 256) {
    int r = i / KPAD, k = i % KPAD;
    int row = row0 + r;
    float a;
    if (MODE == 0) {
      a = (k < K) ? Asrc[(size_t)row * RBF + k] : 0.f;
    } else if (MODE == 1) {
      if (k < 64)        a = atom_ro[(size_t)bidx[2*row]   * UNITS + k];
      else if (k < 128)  a = atom_ro[(size_t)bidx[2*row+1] * UNITS + (k - 64)];
      else               a = bond_feat[(size_t)row * UNITS + (k - 128)];
    } else {
      a = Asrc[(size_t)row * UNITS + k];
    }
    As[i] = f2bf_bits(a);
  }
  __syncthreads();

  const int lane = tid & 31;
  const int wave = tid >> 5;
  const int rt   = wave >> 2;    // row tile 0..1
  const int ct   = wave & 3;     // col tile 0..3
  const int lrow = lane & 15;
  const int hi   = lane >> 4;    // half-wave select

  v8f acc1 = {}; v8f acc2 = {};
  #pragma unroll
  for (int kc = 0; kc < KPAD / 32; ++kc) {
    FragU fa, fb1, fb2;
    // A fragment: lane row = rt*16+lrow; K = kc*32 + hi*16 + {0..15} (contiguous)
    const unsigned short* ap = &As[(rt*16 + lrow) * KPAD + kc*32 + hi*16];
    fa.q[0] = *(const uint4*)(ap);
    fa.q[1] = *(const uint4*)(ap + 8);
    // B fragment: lane col = ct*16+lrow; K = kc*32 + hi*16 + {0..15} (contiguous)
    const unsigned short* bp1 = &Bs1[(ct*16 + lrow) * KPAD + kc*32 + hi*16];
    fb1.q[0] = ((const uint4*)bp1)[0];
    fb1.q[1] = ((const uint4*)bp1)[1];
    const unsigned short* bp2 = &Bs2[(ct*16 + lrow) * KPAD + kc*32 + hi*16];
    fb2.q[0] = ((const uint4*)bp2)[0];
    fb2.q[1] = ((const uint4*)bp2)[1];
    acc1 = __builtin_amdgcn_wmma_f32_16x16x32_bf16(false, fa.v, false, fb1.v,
                                                   (short)0, acc1, false, false);
    acc2 = __builtin_amdgcn_wmma_f32_16x16x32_bf16(false, fa.v, false, fb2.v,
                                                   (short)0, acc2, false, false);
  }

  // epilogue: D layout — vgpr r: M = r + hi*8, N = lrow (within tile)
  const int col = ct*16 + lrow;
  const float b1v = sb1[col], b2v = sb2[col];
  #pragma unroll
  for (int r8 = 0; r8 < 8; ++r8) {
    int row = row0 + rt*16 + hi*8 + r8;
    float g = fused_gate(acc1[r8] + b1v, acc2[r8] + b2v);
    if (MODE == 2) {
      int dst = bidx[2*row + 1];
      atomicAdd(&atom_wr[(size_t)dst * UNITS + col], g);
    } else {
      bond_feat[(size_t)row * UNITS + col] += g;
    }
  }
}

// ---------------- readout ----------------
__global__ void out_kernel(const float* __restrict__ atom_feat,
                           const float* __restrict__ Wout,
                           const float* __restrict__ bout,
                           const int*   __restrict__ batch_atom,
                           float* __restrict__ out) {
  int a = blockIdx.x * blockDim.x + threadIdx.x;
  if (a >= N_ATOMS) return;
  float acc = bout[0];
  #pragma unroll 8
  for (int u = 0; u < UNITS; ++u)
    acc += atom_feat[(size_t)a * UNITS + u] * Wout[u];
  atomicAdd(&out[batch_atom[a]], acc);
}

extern "C" void kernel_launch(void* const* d_in, const int* in_sizes, int n_in,
                              void* d_out, int out_size, void* d_ws, size_t ws_size,
                              hipStream_t stream) {
  const float* atom_positions = (const float*)d_in[0];
  const float* pbc_offsets    = (const float*)d_in[1];
  const float* lattices       = (const float*)d_in[2];
  const float* emb            = (const float*)d_in[3];
  const float* Wb             = (const float*)d_in[4];
  const float* bb             = (const float*)d_in[5];
  const float* Wup            = (const float*)d_in[6];
  const float* bup            = (const float*)d_in[7];
  const float* Wfus1          = (const float*)d_in[8];
  const float* bfus1          = (const float*)d_in[9];
  const float* Wfus2          = (const float*)d_in[10];
  const float* bfus2          = (const float*)d_in[11];
  const float* Wbond1         = (const float*)d_in[12];
  const float* bbond1         = (const float*)d_in[13];
  const float* Wbond2         = (const float*)d_in[14];
  const float* bbond2         = (const float*)d_in[15];
  const float* Watom1         = (const float*)d_in[16];
  const float* batom1         = (const float*)d_in[17];
  const float* Watom2         = (const float*)d_in[18];
  const float* batom2         = (const float*)d_in[19];
  const float* Wout           = (const float*)d_in[20];
  const float* bout           = (const float*)d_in[21];
  const int* atom_types       = (const int*)d_in[22];
  const int* bond_atom_idx    = (const int*)d_in[23];
  const int* triple_idx       = (const int*)d_in[24];
  const int* batch_atom       = (const int*)d_in[25];
  const int* batch_bond       = (const int*)d_in[26];

  char* ws = (char*)d_ws;
  float* vec        = (float*)(ws + 0);                       // 12,000,000 B
  float* rlen       = (float*)(ws + 12000000);                //  4,000,000 B
  float* bond_feat  = (float*)(ws + 16000000);                // 256,000,000 B
  float* agg        = (float*)(ws + 272000000);               // 192,000,000 B
  float* atom_feat  = (float*)(ws + 464000000);               //  12,800,000 B
  float* gate       = (float*)(ws + 476800000);               //   9,600,000 B
  unsigned short* basis = (unsigned short*)(ws + 486400000);  // 144,000,000 B
  int* end_atom     = (int*)(ws + 630400000);                 //   6,000,000 B

  geom_kernel<<<(N_BONDS + 255) / 256, 256, 0, stream>>>(
      atom_positions, pbc_offsets, lattices, bond_atom_idx, batch_bond, vec, rlen);
  bond_init_kernel<<<(N_BONDS * UNITS) / 256, 256, 0, stream>>>(rlen, Wb, bb, bond_feat);
  atom_init_kernel<<<(N_ATOMS * UNITS) / 256, 256, 0, stream>>>(emb, atom_types, atom_feat);
  basis_kernel<<<(N_TRIPLES + 255) / 256, 256, 0, stream>>>(
      vec, triple_idx, bond_atom_idx, basis, end_atom);

  for (int blk = 0; blk < 3; ++blk) {
    gate_kernel<<<(N_ATOMS * RBF) / 256, 256, 0, stream>>>(
        atom_feat, Wup + (size_t)blk * UNITS * RBF, bup + (size_t)blk * RBF, gate);
    hipMemsetAsync(agg, 0, (size_t)N_BONDS * RBF * sizeof(float), stream);
    scatter_kernel<<<(N_TRIPLES * RBF) / 256, 256, 0, stream>>>(
        basis, gate, triple_idx, end_atom, agg);
    dual_gemm_kernel<0, 48, 64><<<N_BONDS / 32, 256, 0, stream>>>(
        agg, nullptr, bond_atom_idx,
        Wfus1 + (size_t)blk * RBF * UNITS, bfus1 + (size_t)blk * UNITS,
        Wfus2 + (size_t)blk * RBF * UNITS, bfus2 + (size_t)blk * UNITS,
        bond_feat, nullptr);
    dual_gemm_kernel<1, 192, 192><<<N_BONDS / 32, 256, 0, stream>>>(
        nullptr, atom_feat, bond_atom_idx,
        Wbond1 + (size_t)blk * 3 * UNITS * UNITS, bbond1 + (size_t)blk * UNITS,
        Wbond2 + (size_t)blk * 3 * UNITS * UNITS, bbond2 + (size_t)blk * UNITS,
        bond_feat, nullptr);
    dual_gemm_kernel<2, 64, 64><<<N_BONDS / 32, 256, 0, stream>>>(
        bond_feat, nullptr, bond_atom_idx,
        Watom1 + (size_t)blk * UNITS * UNITS, batom1 + (size_t)blk * UNITS,
        Watom2 + (size_t)blk * UNITS * UNITS, batom2 + (size_t)blk * UNITS,
        bond_feat, atom_feat);
  }

  hipMemsetAsync(d_out, 0, (size_t)N_GRAPHS * sizeof(float), stream);
  out_kernel<<<(N_ATOMS + 255) / 256, 256, 0, stream>>>(
      atom_feat, Wout, bout, batch_atom, (float*)d_out);
}
